// Head_59133109731970
// MI455X (gfx1250) — compile-verified, matrix operand order checked by reading, not run on hardware
//
#include <hip/hip_runtime.h>
#include <hip/hip_bf16.h>

typedef __attribute__((ext_vector_type(16))) _Float16 v16h;
typedef __attribute__((ext_vector_type(8)))  float    v8f;
typedef unsigned int u32x4 __attribute__((ext_vector_type(4)));
typedef int          i32x8 __attribute__((ext_vector_type(8)));
typedef int          i32x4 __attribute__((ext_vector_type(4)));

#define TDIM 256
#define CDIM 384
#define HDIM 64

#if defined(__has_builtin)
#if __has_builtin(__builtin_amdgcn_tensor_load_to_lds)
#define HAVE_TDM 1
#endif
#endif

__device__ __forceinline__ v8f wmma_f16(v16h a, v16h b, v8f c) {
    // 8 args: (neg_a, A, neg_b, B, c_mod, C, reuse_a, reuse_b)
    return __builtin_amdgcn_wmma_f32_16x16x32_f16(false, a, false, b, (short)0, c, false, false);
}

// Build a v16h A/B operand from two contiguous 8-half LDS segments.
__device__ __forceinline__ v16h lds_pair8(const _Float16* p0, const _Float16* p1) {
    v16h r;
#pragma unroll
    for (int i = 0; i < 8; ++i) { r[i] = p0[i]; r[i + 8] = p1[i]; }
    return r;
}

#ifdef HAVE_TDM
// Issue a TDM 2D tile load: rows x cols f32, row-major (stride == cols),
// global -> LDS at byte offset lds_off. D# layout per CDNA5 ISA 8.3/8.4.
__device__ __forceinline__ void tdm_load_2d_f32(const float* gsrc, unsigned lds_off,
                                                int rows, int cols) {
    const unsigned long long ga = (unsigned long long)(uintptr_t)gsrc;
    u32x4 g0;
    g0[0] = 1u;                                        // count=1, user descriptor
    g0[1] = lds_off;                                   // lds_addr[31:0]
    g0[2] = (unsigned)(ga & 0xFFFFFFFFu);              // global_addr[31:0]
    g0[3] = (unsigned)((ga >> 32) & 0x1FFFFFFu)        // global_addr[56:32]
          | 0x80000000u;                               // type=2 ("image")
    i32x8 g1;
    g1[0] = 0x00020000;                                // data_size=4B; no flags
    g1[1] = (cols & 0xFFFF) << 16;                     // tensor_dim0[15:0]
    g1[2] = (rows & 0xFFFF) << 16;                     // dim0 hi=0 | tensor_dim1[15:0]
    g1[3] = (cols & 0xFFFF) << 16;                     // dim1 hi=0 | tile_dim0
    g1[4] = (rows & 0xFFFF);                           // tile_dim1 | tile_dim2=0
    g1[5] = cols;                                      // tensor_dim0_stride[31:0]
    g1[6] = 0;
    g1[7] = 0;
    const i32x4 z4 = {0, 0, 0, 0};
#if __clang_major__ >= 23
    const i32x8 z8 = {0, 0, 0, 0, 0, 0, 0, 0};
    __builtin_amdgcn_tensor_load_to_lds(g0, g1, z4, z4, z8, 0);
#else
    __builtin_amdgcn_tensor_load_to_lds(g0, g1, z4, z4, 0);
#endif
#if __has_builtin(__builtin_amdgcn_s_wait_tensorcnt)
    __builtin_amdgcn_s_wait_tensorcnt(0);
#else
    asm volatile("s_wait_tensorcnt 0x0" ::: "memory");
#endif
}
#endif

__global__ __launch_bounds__(256, 1)
void attn_fused(const float* __restrict__ x,
                const float* __restrict__ Wq,
                const float* __restrict__ Wk,
                const float* __restrict__ Wv,
                float* __restrict__ out) {
    // W^T (q|k|v concatenated, col-major, padded): 150,528 B
    __shared__ _Float16 sWT[192][392];
    // Pool aliased as: phase-0 TDM f32 staging (49,152 B needed) OR
    // { sK[256][72] | sVT[64][264] | sScr[8][1024] } = 87,040 B
    __shared__ __align__(16) unsigned char sPool[36864 + 33792 + 16384];
    _Float16 (*sK)[72]    = (_Float16(*)[72])  (sPool);
    _Float16 (*sVT)[264]  = (_Float16(*)[264]) (sPool + 36864);
    _Float16 (*sScr)[1024]= (_Float16(*)[1024])(sPool + 36864 + 33792);
    float* wstage         = (float*)sPool;

    const int tid  = threadIdx.x;
    const int w    = tid >> 5;       // wave id (0..7), wave32
    const int lane = tid & 31;
    const int hl   = lane >> 4;      // half-wave (K-range select in A/B layouts)
    const int ln   = lane & 15;
    const int b    = blockIdx.x;

    // ---- Phase 0: stage W^T (f16) into LDS; cols 0..63=Wq, 64..127=Wk, 128..191=Wv
#ifdef HAVE_TDM
    // TDM path: DMA each W half-matrix (192x64 f32) into LDS, then convert/transpose.
    for (int mat = 0; mat < 3; ++mat) {
        const float* Wm = (mat == 0) ? Wq : (mat == 1) ? Wk : Wv;
#pragma unroll 1
        for (int half = 0; half < 2; ++half) {
            if (w == 0) {  // wave-uniform: one wave issues the DMA, waits tensorcnt
                tdm_load_2d_f32(Wm + half * 192 * HDIM,
                                (unsigned)(uintptr_t)wstage, 192, HDIM);
            }
            __syncthreads();                     // publish DMA'd data to all waves
            for (int e = tid; e < 192 * HDIM; e += 256) {
                const int k = e >> 6, h = e & 63;
                sWT[mat * 64 + h][half * 192 + k] = (_Float16)wstage[e];
            }
            __syncthreads();                     // wstage free for reuse
        }
    }
#else
    for (int e = tid; e < 192 * CDIM; e += 256) {
        const int col = e / CDIM;
        const int k   = e - col * CDIM;
        const float* W = (col < 64) ? Wq : (col < 128) ? Wk : Wv;
        sWT[col][k] = (_Float16)W[k * HDIM + (col & 63)];
    }
    __syncthreads();
#endif

    const float* xb = x + (size_t)b * TDIM * CDIM;

    v16h qa[2][2];   // this wave's two Q tiles, A-layout (16x32 halves each)

    // ---- Phase 1: QKV = x @ [Wq|Wk|Wv]; wave w owns row-tiles 2w, 2w+1
#pragma unroll 1
    for (int j = 0; j < 2; ++j) {
        const int rt = 2 * w + j;
        v8f acc[12];
#pragma unroll
        for (int c = 0; c < 12; ++c) acc[c] = {};

        const float* xr = xb + (size_t)(16 * rt + ln) * CDIM;

#pragma unroll 1
        for (int kc = 0; kc < 12; ++kc) {
            const int kb = kc * 32 + hl * 8;        // per-half K base (A layout)
            float4 f0 = *(const float4*)(xr + kb);
            float4 f1 = *(const float4*)(xr + kb + 4);
            float4 f2 = *(const float4*)(xr + kb + 16);
            float4 f3 = *(const float4*)(xr + kb + 20);
            if (kc < 11) __builtin_prefetch(xr + kb + 32, 0, 1);  // global_prefetch_b8
            v16h a;
            a[0]=(_Float16)f0.x; a[1]=(_Float16)f0.y; a[2]=(_Float16)f0.z; a[3]=(_Float16)f0.w;
            a[4]=(_Float16)f1.x; a[5]=(_Float16)f1.y; a[6]=(_Float16)f1.z; a[7]=(_Float16)f1.w;
            a[8]=(_Float16)f2.x; a[9]=(_Float16)f2.y; a[10]=(_Float16)f2.z; a[11]=(_Float16)f2.w;
            a[12]=(_Float16)f3.x; a[13]=(_Float16)f3.y; a[14]=(_Float16)f3.z; a[15]=(_Float16)f3.w;
#pragma unroll
            for (int c = 0; c < 12; ++c) {
                const int col = c * 16 + ln;        // B layout: lane -> output column
                v16h bm = lds_pair8(&sWT[col][kb], &sWT[col][kb + 16]);
                acc[c] = wmma_f16(a, bm, acc[c]);
            }
        }

        // Scatter results: c0..3 -> Q scratch, c4..7 -> K rows, c8..11 -> V^T
#pragma unroll
        for (int c = 0; c < 12; ++c) {
#pragma unroll
            for (int r = 0; r < 8; ++r) {
                const int m = hl ? (r + 8) : r;     // C/D layout row mapping
                const _Float16 hv = (_Float16)acc[c][r];
                if (c < 4)      sScr[w][m * 64 + c * 16 + ln] = hv;
                else if (c < 8) sK[16 * rt + m][(c - 4) * 16 + ln] = hv;
                else            sVT[(c - 8) * 16 + ln][16 * rt + m] = hv;
            }
        }
        // Re-load this Q tile in A layout (16x64 -> two 16x32 operands)
        {
            const int kb0 = hl * 8;
            const _Float16* qs = &sScr[w][ln * 64];
            qa[j][0] = lds_pair8(qs + kb0,      qs + kb0 + 16);
            qa[j][1] = lds_pair8(qs + 32 + kb0, qs + 32 + kb0 + 16);
        }
    }
    __syncthreads();   // K/V complete for the whole batch

    // ---- Phase 2: causal flash attention, 32-wide key blocks
    const float scale = 0.125f;   // 1/sqrt(64)
#pragma unroll 1
    for (int j = 0; j < 2; ++j) {
        const int qi    = 2 * w + j;
        const int qrow0 = 16 * qi;
        v8f o[4];
#pragma unroll
        for (int ct = 0; ct < 4; ++ct) o[ct] = {};
        float mrun[8], lrun[8];
#pragma unroll
        for (int r = 0; r < 8; ++r) { mrun[r] = -3.0e38f; lrun[r] = 0.0f; }

        const int nkb = (qi >> 1) + 1;              // causal: only blocks <= diagonal
#pragma unroll 1
        for (int kb = 0; kb < nkb; ++kb) {
            const int base = 32 * kb;
            v8f s0 = {}, s1 = {};
#pragma unroll
            for (int ch = 0; ch < 2; ++ch) {        // K-dim = H = 64 = 2 chunks of 32
                const int hk = ch * 32 + hl * 8;
                v16h bk0 = lds_pair8(&sK[base + ln][hk],      &sK[base + ln][hk + 16]);
                v16h bk1 = lds_pair8(&sK[base + 16 + ln][hk], &sK[base + 16 + ln][hk + 16]);
                s0 = wmma_f16(qa[j][ch], bk0, s0);
                s1 = wmma_f16(qa[j][ch], bk1, s1);
            }
            // Online softmax (row = (vgpr r, half); reduce across 16 lanes of the half)
            _Float16 p0h[8], p1h[8]; float alpha[8];
#pragma unroll
            for (int r = 0; r < 8; ++r) {
                const int m    = hl ? (r + 8) : r;
                const int grow = qrow0 + m;
                const int c0   = base + ln;
                float e0 = (c0      <= grow) ? s0[r] * scale : -3.0e38f;
                float e1 = (c0 + 16 <= grow) ? s1[r] * scale : -3.0e38f;
                float mx = fmaxf(e0, e1);
                mx = fmaxf(mx, __shfl_xor(mx, 1, 16));
                mx = fmaxf(mx, __shfl_xor(mx, 2, 16));
                mx = fmaxf(mx, __shfl_xor(mx, 4, 16));
                mx = fmaxf(mx, __shfl_xor(mx, 8, 16));
                const float mnew = fmaxf(mrun[r], mx);
                const float al = __expf(mrun[r] - mnew);
                const float p0 = __expf(e0 - mnew);
                const float p1 = __expf(e1 - mnew);
                float rs = p0 + p1;
                rs += __shfl_xor(rs, 1, 16);
                rs += __shfl_xor(rs, 2, 16);
                rs += __shfl_xor(rs, 4, 16);
                rs += __shfl_xor(rs, 8, 16);
                lrun[r] = lrun[r] * al + rs;
                mrun[r] = mnew;
                alpha[r] = al;
                p0h[r] = (_Float16)p0; p1h[r] = (_Float16)p1;
            }
#pragma unroll
            for (int ct = 0; ct < 4; ++ct)
#pragma unroll
                for (int r = 0; r < 8; ++r) o[ct][r] *= alpha[r];

            // P (C-layout f32) -> wave scratch f16 -> A-layout operand
#pragma unroll
            for (int r = 0; r < 8; ++r) {
                const int m = hl ? (r + 8) : r;
                sScr[w][m * 32 + ln]      = p0h[r];
                sScr[w][m * 32 + 16 + ln] = p1h[r];
            }
            v16h pa = lds_pair8(&sScr[w][ln * 32 + hl * 8],
                                &sScr[w][ln * 32 + hl * 8 + 16]);
#pragma unroll
            for (int ct = 0; ct < 4; ++ct) {
                const _Float16* vr = &sVT[ct * 16 + ln][base + hl * 8];
                v16h bv = lds_pair8(vr, vr + 16);
                o[ct] = wmma_f16(pa, bv, o[ct]);
            }
        }
        // Epilogue: O / l, f32 store (lanes 0..15 consecutive cols -> coalesced)
        float* ob = out + ((size_t)b * TDIM + qrow0) * HDIM;
#pragma unroll
        for (int r = 0; r < 8; ++r) {
            const int m = hl ? (r + 8) : r;
            const float inv = 1.0f / lrun[r];
#pragma unroll
            for (int ct = 0; ct < 4; ++ct)
                ob[m * HDIM + ct * 16 + ln] = o[ct][r] * inv;
        }
    }
}

extern "C" void kernel_launch(void* const* d_in, const int* in_sizes, int n_in,
                              void* d_out, int out_size, void* d_ws, size_t ws_size,
                              hipStream_t stream) {
    const float* x  = (const float*)d_in[0];
    const float* Wq = (const float*)d_in[1];
    const float* Wk = (const float*)d_in[2];
    const float* Wv = (const float*)d_in[3];
    float* out = (float*)d_out;
    attn_fused<<<dim3(256), dim3(256), 0, stream>>>(x, Wq, Wk, Wv, out);
}
